// FoldingBlock_77841987273310
// MI455X (gfx1250) — matrix operand order repaired and checked by reading, not compiled, not run on hardware
//
#include <hip/hip_runtime.h>

typedef _Float16 half_t;
typedef __attribute__((ext_vector_type(16))) _Float16 v16h;
typedef __attribute__((ext_vector_type(8)))  _Float16 v8h;
typedef __attribute__((ext_vector_type(8)))  float    v8f;

// ---------------------------------------------------------------------------
// CDNA5 async memory->LDS copy (ASYNCcnt-tracked, no VGPR round trip)
// ---------------------------------------------------------------------------
__device__ __forceinline__ void async_g2l_b128(unsigned lds_addr, const void* gaddr) {
  asm volatile("global_load_async_to_lds_b128 %0, %1, off"
               :: "v"(lds_addr), "v"(gaddr) : "memory");
}
__device__ __forceinline__ void wait_async0() {
  asm volatile("s_wait_asynccnt 0x0" ::: "memory");
}

// ---------------------------------------------------------------------------
// Epilogue modes for the fused WMMA GEMM
// ---------------------------------------------------------------------------
enum EpiMode {
  EPI_F16        = 0,  // C0(f16) = relu(acc+bias)
  EPI_F16T       = 1,  // C0(f16) transposed store: C0[n*ldc+m]
  EPI_F32        = 2,  // C0(f32) = acc               (attention scores)
  EPI_F32_F16    = 3,  // C0(f32) = v; C1(f16) = v    (up1 out + down input)
  EPI_ADD_F32    = 4,  // C0(f32) += v                (p1 + p2u)
  EPI_ADDSRC_F16 = 5,  // C0(f16) = acc + S(f16)      (residual x = pcat+applied)
  EPI_DELTA_F16  = 6,  // C0(f16) = S(f32) - relu(acc+bias)   (p - down)
};

static constexpr int kLds = 40;   // LDS row stride in halves: 32 data + 8 pad (80B, conflict-free)

// ---------------------------------------------------------------------------
// Fused batched GEMM:  C = epilogue(A(MxK,f16) * B(KxN) + bias)
// B is supplied as BT: row-major (N x K) f16, so LDS staging and B-fragment
// loads are contiguous. Shapes must divide the tile grid exactly.
// Staging uses GLOBAL_LOAD_ASYNC_TO_LDS_B128 double-buffered against the
// WMMA compute; handoff = s_wait_asynccnt 0 (own wave) + s_barrier (workgroup).
// sched_barrier(0) pins the fragment ds_loads as one issue batch ahead of the
// WMMA stream so dscnt waits cascade instead of fully draining per pair.
// ---------------------------------------------------------------------------
template<int BM, int BN, int EPI, bool RELU, bool BIAS>
__global__ __launch_bounds__(256)
void wmma_gemm(const half_t* __restrict__ A, long long strideA, int lda,
               const half_t* __restrict__ BT, long long strideBT, int ldbt,
               const float* __restrict__ bias,
               void* __restrict__ C0, long long strideC0, int ldc0,
               void* __restrict__ C1, long long strideC1,
               const void* __restrict__ S, long long strideS, int ldS,
               int K)
{
  static_assert(BM == 64 || BM == 128, "BM must be 64 or 128");
  constexpr int WMv = BM / 32;      // waves along M
  constexpr int WNv = 8 / WMv;      // waves along N
  constexpr int WNT = BN / WNv;     // wave tile N
  constexpr int FN  = WNT / 16;     // 16x16 fragments along N per wave
  constexpr int AV  = (BM * 4) / 256;  // async b128 issues per thread (A tile)
  constexpr int BV  = (BN * 4) / 256;  // async b128 issues per thread (B tile)

  __shared__ half_t As[2][BM * kLds];
  __shared__ half_t Bs[2][BN * kLds];

  const int tid  = threadIdx.x;
  const int lane = tid & 31;
  const int wid  = tid >> 5;
  const int wm   = wid % WMv;
  const int wn   = wid / WMv;
  const int llo  = lane & 15;
  const int lhi  = lane >> 4;
  const int bz   = blockIdx.z;
  const int bm0  = blockIdx.y * BM;
  const int bn0  = blockIdx.x * BN;

  const half_t* Ab = A  + (long long)bz * strideA  + (long long)bm0 * lda;
  const half_t* Bb = BT + (long long)bz * strideBT + (long long)bn0 * ldbt;

  v8f acc[2][FN];
  for (int i = 0; i < 2; ++i)
    for (int j = 0; j < FN; ++j) {
      v8f z = {0.f, 0.f, 0.f, 0.f, 0.f, 0.f, 0.f, 0.f};
      acc[i][j] = z;
    }

  auto stage = [&](int buf, int k0) {
    #pragma unroll
    for (int i = 0; i < AV; ++i) {
      const int idx = tid + i * 256;
      const int r = idx >> 2, sg = idx & 3;
      unsigned dst = (unsigned)(unsigned long long)(void*)&As[buf][r * kLds + sg * 8];
      async_g2l_b128(dst, Ab + (long long)r * lda + (k0 + sg * 8));
    }
    #pragma unroll
    for (int i = 0; i < BV; ++i) {
      const int idx = tid + i * 256;
      const int r = idx >> 2, sg = idx & 3;
      unsigned dst = (unsigned)(unsigned long long)(void*)&Bs[buf][r * kLds + sg * 8];
      async_g2l_b128(dst, Bb + (long long)r * ldbt + (k0 + sg * 8));
    }
  };

  stage(0, 0);
  wait_async0();
  __syncthreads();

  const int nk = K >> 5;
  for (int kt = 0; kt < nk; ++kt) {
    const int cur = kt & 1;
    const bool more = (kt + 1 < nk);
    if (more) stage(cur ^ 1, (kt + 1) << 5);   // async DMA overlaps compute below

    const half_t* as = As[cur];
    const half_t* bs = Bs[cur];

    // A fragments (16x32 f16): lane llo = row, half-wave selects K 0-7/16-23 vs 8-15/24-31
    v16h af[2];
    #pragma unroll
    for (int fm = 0; fm < 2; ++fm) {
      const half_t* pa = as + (wm * 32 + fm * 16 + llo) * kLds + lhi * 8;
      v8h lo = *(const v8h*)(pa);
      v8h hi = *(const v8h*)(pa + 16);
      af[fm] = __builtin_shufflevector(lo, hi, 0,1,2,3,4,5,6,7,8,9,10,11,12,13,14,15);
    }
    // B fragments (32x16 f16): lane llo = column, half-wave selects K 0-7/16-23 vs 8-15/24-31
    v16h bf[FN];
    #pragma unroll
    for (int fn = 0; fn < FN; ++fn) {
      const half_t* pb = bs + (wn * WNT + fn * 16 + llo) * kLds + lhi * 8;
      v8h lo = *(const v8h*)(pb);
      v8h hi = *(const v8h*)(pb + 16);
      bf[fn] = __builtin_shufflevector(lo, hi, 0,1,2,3,4,5,6,7,8,9,10,11,12,13,14,15);
    }

    // Scheduling fence: keep all ds_loads above as one issue batch; WMMAs below
    // then drain them with partial dscnt waits while later loads are in flight.
    __builtin_amdgcn_sched_barrier(0);

    #pragma unroll
    for (int fn = 0; fn < FN; ++fn) {
      #pragma unroll
      for (int fm = 0; fm < 2; ++fm) {
        acc[fm][fn] = __builtin_amdgcn_wmma_f32_16x16x32_f16(
            false, af[fm], false, bf[fn], (short)0, acc[fm][fn], false, false);
      }
    }
    __builtin_amdgcn_sched_barrier(0);

    if (more) wait_async0();   // my next-buffer DMAs have landed in LDS
    __syncthreads();           // everyone's have
  }

  // Epilogue: accumulator VGPR v holds row (fragBase + v + 8*lhi), column llo
  #pragma unroll
  for (int fm = 0; fm < 2; ++fm) {
    #pragma unroll
    for (int fn = 0; fn < FN; ++fn) {
      const int gm0 = bm0 + wm * 32 + fm * 16 + lhi * 8;
      const int gn  = bn0 + wn * WNT + fn * 16 + llo;
      float bv = 0.f;
      if constexpr (BIAS) bv = bias[gn];
      float vals[8];
      #pragma unroll
      for (int v = 0; v < 8; ++v) {
        float x = acc[fm][fn][v] + bv;
        if constexpr (RELU) x = (x > 0.f) ? x : 0.f;
        vals[v] = x;
      }
      if constexpr (EPI == EPI_F16T) {
        // 8 consecutive rows per lane -> one packed 16B transposed store
        v8h pk;
        #pragma unroll
        for (int v = 0; v < 8; ++v) pk[v] = (half_t)vals[v];
        *(v8h*)((half_t*)C0 + (long long)bz * strideC0 + (long long)gn * ldc0 + gm0) = pk;
      } else {
        #pragma unroll
        for (int v = 0; v < 8; ++v) {
          const long long row = (long long)(gm0 + v);
          const long long ic  = (long long)bz * strideC0 + row * ldc0 + gn;
          if constexpr (EPI == EPI_F16) {
            ((half_t*)C0)[ic] = (half_t)vals[v];
          } else if constexpr (EPI == EPI_F32) {
            ((float*)C0)[ic] = vals[v];
          } else if constexpr (EPI == EPI_F32_F16) {
            ((float*)C0)[ic] = vals[v];
            ((half_t*)C1)[(long long)bz * strideC1 + row * ldc0 + gn] = (half_t)vals[v];
          } else if constexpr (EPI == EPI_ADD_F32) {
            ((float*)C0)[ic] += vals[v];
          } else if constexpr (EPI == EPI_ADDSRC_F16) {
            float sv = (float)((const half_t*)S)[(long long)bz * strideS + row * ldS + gn];
            ((half_t*)C0)[ic] = (half_t)(vals[v] + sv);
          } else if constexpr (EPI == EPI_DELTA_F16) {
            float sv = ((const float*)S)[(long long)bz * strideS + row * ldS + gn];
            ((half_t*)C0)[ic] = (half_t)(sv - vals[v]);
          }
        }
      }
    }
  }
}

// ---------------------------------------------------------------------------
// Build A_up (256 x 544, zero-padded 514->544) and pcat[:, 0:512] = tile(p)
// ---------------------------------------------------------------------------
template<typename SrcT>
__global__ __launch_bounds__(256)
void prep_up(const SrcT* __restrict__ src, long long strideSrc,
             const float* __restrict__ pts,
             half_t* __restrict__ Aup, long long strideAup,
             half_t* __restrict__ pcat, long long stridePcat)
{
  const int b = blockIdx.z;
  const int r = blockIdx.x;            // 0..255
  const int t = threadIdx.x;
  const SrcT* s = src + (long long)b * strideSrc + (long long)(r & 63) * 512;
  half_t* arow = Aup  + (long long)b * strideAup  + (long long)r * 544;
  half_t* prow = pcat + (long long)b * stridePcat + (long long)r * 1024;
  for (int c = t; c < 512; c += 256) {
    half_t v = (half_t)(float)s[c];
    arow[c] = v;
    prow[c] = v;
  }
  if (t < 32) {
    half_t v = (half_t)0.f;
    if (t < 2) v = (half_t)pts[r * 2 + t];
    arow[512 + t] = v;
  }
}

// ---------------------------------------------------------------------------
// Row softmax over 256 columns, f32 in -> f16 out
// ---------------------------------------------------------------------------
__global__ __launch_bounds__(256)
void softmax256(const float* __restrict__ mm, half_t* __restrict__ attn,
                long long strideIn, long long strideOut)
{
  const int b = blockIdx.y, r = blockIdx.x, t = threadIdx.x;
  const float* x = mm + (long long)b * strideIn + (long long)r * 256;
  __shared__ float red[256];
  float v = x[t];
  red[t] = v; __syncthreads();
  for (int s = 128; s > 0; s >>= 1) { if (t < s) red[t] = fmaxf(red[t], red[t + s]); __syncthreads(); }
  float mx = red[0]; __syncthreads();
  float e = __expf(v - mx);
  red[t] = e; __syncthreads();
  for (int s = 128; s > 0; s >>= 1) { if (t < s) red[t] += red[t + s]; __syncthreads(); }
  float inv = 1.f / red[0];
  attn[(long long)b * strideOut + (long long)r * 256 + t] = (half_t)(e * inv);
}

// ---------------------------------------------------------------------------
// Pack weight W (K x N, f32, row-major) -> WT (N x Kp, f16) with zero pad k>=K
// ---------------------------------------------------------------------------
__global__ __launch_bounds__(256)
void pack_wt(const float* __restrict__ W, half_t* __restrict__ WT, int K, int N, int Kp)
{
  long long i = (long long)blockIdx.x * 256 + threadIdx.x;
  if (i >= (long long)N * Kp) return;
  int n = (int)(i / Kp);
  int k = (int)(i - (long long)n * Kp);
  WT[i] = (k < K) ? (half_t)W[(long long)k * N + n] : (half_t)0.f;
}

// ---------------------------------------------------------------------------
// Reproduce SANet.sample_2D grid (m*n x 2)
// ---------------------------------------------------------------------------
__global__ void compute_pts(const int* __restrict__ pm, const int* __restrict__ pn,
                            float* __restrict__ pts)
{
  int t = threadIdx.x;
  int m = *pm, n = *pn;
  if (t < m * n) {
    int a = t / m, c = t % m;
    float iy = rintf((float)a * 45.f / (float)(n - 1));   // numpy round (ties-even)
    float ix = rintf((float)c * 45.f / (float)(m - 1));
    pts[2 * t + 0] = -0.3f + 0.6f * (iy / 45.f);          // xs[iy[t/m]]
    pts[2 * t + 1] = -0.3f + 0.6f * (ix / 45.f);          // ys[ix[t%m]]
  }
}

// ---------------------------------------------------------------------------
// Host launcher helpers
// ---------------------------------------------------------------------------
template<int BM, int BN, int EPI, bool RELU, bool BIAS>
static inline void launch_gemm(hipStream_t stream, int cb, int M, int N, int K,
                               const half_t* A, long long sA, int lda,
                               const half_t* BT, long long sBT, int ldbt,
                               const float* bias,
                               void* C0, long long sC0, int ldc0,
                               void* C1, long long sC1,
                               const void* S, long long sS, int ldS)
{
  dim3 grid(N / BN, M / BM, cb);
  wmma_gemm<BM, BN, EPI, RELU, BIAS><<<grid, dim3(256), 0, stream>>>(
      A, sA, lda, BT, sBT, ldbt, bias, C0, sC0, ldc0, C1, sC1, S, sS, ldS, K);
}

extern "C" void kernel_launch(void* const* d_in, const int* in_sizes, int n_in,
                              void* d_out, int out_size, void* d_ws, size_t ws_size,
                              hipStream_t stream)
{
  (void)n_in; (void)out_size;
  const float* p      = (const float*)d_in[0];
  const float* W_up   = (const float*)d_in[1];
  const float* b_up   = (const float*)d_in[2];
  const float* Wh     = (const float*)d_in[3];
  const float* bh     = (const float*)d_in[4];
  const float* Wl     = (const float*)d_in[5];
  const float* bl     = (const float*)d_in[6];
  const float* Wg     = (const float*)d_in[7];
  const float* bg     = (const float*)d_in[8];
  const float* Wf1    = (const float*)d_in[9];
  const float* bf1    = (const float*)d_in[10];
  const float* bWf2   = (const float*)d_in[11];
  const float* bf2    = (const float*)d_in[12];
  const float* W_down = (const float*)d_in[13];
  const float* b_down = (const float*)d_in[14];
  const int*   dm     = (const int*)d_in[15];
  const int*   dn     = (const int*)d_in[16];
  float* out = (float*)d_out;

  const int B = in_sizes[0] / (64 * 512);

  // ---- workspace carve (bump allocator, 256B aligned) ----
  char* base = (char*)d_ws;
  size_t off = 0;
  auto alloc = [&](size_t bytes) -> char* {
    char* r = base + off;
    off = (off + bytes + 255) & ~(size_t)255;
    return r;
  };

  // fixed: f16 transposed/padded weights + pts
  half_t* WTup   = (half_t*)alloc((size_t)512  * 544  * 2);
  half_t* WTh    = (half_t*)alloc((size_t)256  * 1024 * 2);
  half_t* WTl    = (half_t*)alloc((size_t)256  * 1024 * 2);
  half_t* WTg    = (half_t*)alloc((size_t)1024 * 1024 * 2);
  half_t* WTf1   = (half_t*)alloc((size_t)512  * 1024 * 2);
  half_t* WTf2   = (half_t*)alloc((size_t)256  * 512  * 2);
  half_t* WTdown = (half_t*)alloc((size_t)512  * 1024 * 2);
  float*  ptsbuf = (float*)alloc((size_t)256 * 2 * 4);

  // per-batch activation strides (elements)
  const long long AUP_ST = 256 * 544, PCAT_ST = 256 * 1024, HB_ST = 256 * 256;
  const long long GT_ST = 1024 * 256, MM_ST = 256 * 256, ATT_ST = 256 * 256;
  const long long F1_ST = 256 * 512, ADN_ST = 64 * 1024, PD_ST = 64 * 512;
  const size_t perBatch =
      (size_t)(AUP_ST + PCAT_ST + HB_ST * 2 + GT_ST + ATT_ST + F1_ST + ADN_ST + PD_ST) * 2
      + (size_t)MM_ST * 4 + 16 * 256 /* alignment slack */;

  size_t avail = ws_size > off ? ws_size - off : perBatch;
  int CH = (int)(avail / perBatch);
  if (CH < 1) CH = 1;
  if (CH > B) CH = B;

  half_t* Aup    = (half_t*)alloc((size_t)CH * AUP_ST * 2);
  half_t* pcat   = (half_t*)alloc((size_t)CH * PCAT_ST * 2);
  half_t* hbuf   = (half_t*)alloc((size_t)CH * HB_ST * 2);
  half_t* lbuf   = (half_t*)alloc((size_t)CH * HB_ST * 2);
  half_t* gT     = (half_t*)alloc((size_t)CH * GT_ST * 2);
  float*  mmb    = (float*)alloc((size_t)CH * MM_ST * 4);
  half_t* attn   = (half_t*)alloc((size_t)CH * ATT_ST * 2);
  half_t* f1b    = (half_t*)alloc((size_t)CH * F1_ST * 2);
  half_t* adown  = (half_t*)alloc((size_t)CH * ADN_ST * 2);
  half_t* pdelta = (half_t*)alloc((size_t)CH * PD_ST * 2);

  // ---- one-time (per call) weight packing + pts ----
  auto packW = [&](const float* W, half_t* WT, int K, int N, int Kp) {
    long long total = (long long)N * Kp;
    pack_wt<<<dim3((unsigned)((total + 255) / 256)), dim3(256), 0, stream>>>(W, WT, K, N, Kp);
  };
  packW(W_up,   WTup,   514,  512,  544);
  packW(Wh,     WTh,    1024, 256,  1024);
  packW(Wl,     WTl,    1024, 256,  1024);
  packW(Wg,     WTg,    1024, 1024, 1024);
  packW(Wf1,    WTf1,   1024, 512,  1024);
  packW(bWf2,   WTf2,   512,  256,  512);
  packW(W_down, WTdown, 1024, 512,  1024);
  compute_pts<<<dim3(1), dim3(256), 0, stream>>>(dm, dn, ptsbuf);

  // ---- chunked pipeline over batches ----
  for (int b0 = 0; b0 < B; b0 += CH) {
    const int cb = (B - b0 < CH) ? (B - b0) : CH;

    auto run_up = [&](int pass) {
      // p2d = relu(A_up @ W_up + b) -> pcat[:,512:1024]
      launch_gemm<128,128,EPI_F16,true,true>(stream, cb, 256, 512, 544,
          Aup, AUP_ST, 544, WTup, 0, 544, b_up,
          pcat + 512, PCAT_ST, 1024, nullptr, 0, nullptr, 0, 0);
      // h, l
      launch_gemm<128,128,EPI_F16,true,true>(stream, cb, 256, 256, 1024,
          pcat, PCAT_ST, 1024, WTh, 0, 1024, bh,
          hbuf, HB_ST, 256, nullptr, 0, nullptr, 0, 0);
      launch_gemm<128,128,EPI_F16,true,true>(stream, cb, 256, 256, 1024,
          pcat, PCAT_ST, 1024, WTl, 0, 1024, bl,
          lbuf, HB_ST, 256, nullptr, 0, nullptr, 0, 0);
      // g, stored transposed (1024 x 256) so it is BT for the apply GEMM
      launch_gemm<128,128,EPI_F16T,true,true>(stream, cb, 256, 1024, 1024,
          pcat, PCAT_ST, 1024, WTg, 0, 1024, bg,
          gT, GT_ST, 256, nullptr, 0, nullptr, 0, 0);
      // mm = l @ h^T  (h row-major IS BT here)
      launch_gemm<128,128,EPI_F32,false,false>(stream, cb, 256, 256, 256,
          lbuf, HB_ST, 256, hbuf, HB_ST, 256, nullptr,
          mmb, MM_ST, 256, nullptr, 0, nullptr, 0, 0);
      // softmax -> attn (f16)
      softmax256<<<dim3(256, cb), dim3(256), 0, stream>>>(mmb, attn, MM_ST, ATT_ST);
      // x = pcat + attn @ g  (in-place residual into pcat)
      launch_gemm<128,128,EPI_ADDSRC_F16,false,false>(stream, cb, 256, 1024, 256,
          attn, ATT_ST, 256, gT, GT_ST, 256, nullptr,
          pcat, PCAT_ST, 1024, nullptr, 0, pcat, PCAT_ST, 1024);
      // f1 = relu(x @ Wf1 + b)
      launch_gemm<128,128,EPI_F16,true,true>(stream, cb, 256, 512, 1024,
          pcat, PCAT_ST, 1024, WTf1, 0, 1024, bf1,
          f1b, F1_ST, 512, nullptr, 0, nullptr, 0, 0);
      // up_out = relu(f1 @ Wf2 + b)
      if (pass == 1) {
        // write p1 -> d_out (f32) and f16 copy -> adown (the (64,1024) reshape is a flat view)
        launch_gemm<128,128,EPI_F32_F16,true,true>(stream, cb, 256, 256, 512,
            f1b, F1_ST, 512, WTf2, 0, 512, bf2,
            out + (long long)b0 * 65536, 65536, 256, adown, ADN_ST, nullptr, 0, 0);
      } else {
        // d_out += p2u
        launch_gemm<128,128,EPI_ADD_F32,true,true>(stream, cb, 256, 256, 512,
            f1b, F1_ST, 512, WTf2, 0, 512, bf2,
            out + (long long)b0 * 65536, 65536, 256, nullptr, 0, nullptr, 0, 0);
      }
    };

    // pass 1: up(p)
    prep_up<float><<<dim3(256, 1, cb), dim3(256), 0, stream>>>(
        p + (long long)b0 * 64 * 512, (long long)64 * 512, ptsbuf,
        Aup, AUP_ST, pcat, PCAT_ST);
    run_up(1);

    // p_delta = p - relu(p1.reshape(64,1024) @ W_down + b)
    launch_gemm<64,128,EPI_DELTA_F16,true,true>(stream, cb, 64, 512, 1024,
        adown, ADN_ST, 1024, WTdown, 0, 1024, b_down,
        pdelta, PD_ST, 512, nullptr, 0,
        p + (long long)b0 * 64 * 512, (long long)64 * 512, 512);

    // pass 2: up(p_delta), accumulate into d_out
    prep_up<half_t><<<dim3(256, 1, cb), dim3(256), 0, stream>>>(
        pdelta, PD_ST, ptsbuf, Aup, AUP_ST, pcat, PCAT_ST);
    run_up(2);
  }
}
